// Attention_2594160247350
// MI455X (gfx1250) — matrix operand order repaired and checked by reading, not compiled, not run on hardware
//
#include <hip/hip_runtime.h>

// MI455X / gfx1250: wave32, WMMA 16x16x32 f16 (fp32 accumulate).
typedef __attribute__((ext_vector_type(16))) _Float16 v16h;
typedef __attribute__((ext_vector_type(8)))  float    v8f;

#define B_ 8
#define T_ 2048
#define D_ 512
#define H_ 64

// ---------------------------------------------------------------------------
// Kernel 1: fused QKV projection.  q,k stored row-major [B*T][H] f16;
// v stored TRANSPOSED [B][H][T] f16 so the attention kernel's V B-fragments
// are contiguous (t-dimension per lane).
// One wave = one 16-row strip, all 4 column tiles of one projection.
// ---------------------------------------------------------------------------
__global__ __launch_bounds__(256) void qkv_proj_kernel(
    const float* __restrict__ x,
    const float* __restrict__ Wq, const float* __restrict__ Wk,
    const float* __restrict__ Wv,
    _Float16* __restrict__ qkv)            // [3][B*T*H] f16
{
    __shared__ unsigned lds_w[1024];       // 4 ntiles x 32 lanes x 8 dwords (f16 pairs)

    const int tid  = threadIdx.x;
    const int wave = tid >> 5;
    const int lane = tid & 31;
    const int half = lane >> 4;
    const int l15  = lane & 15;

    const int proj = blockIdx.y;
    const float* W = (proj == 0) ? Wq : ((proj == 1) ? Wk : Wv);
    _Float16* out  = qkv + (size_t)proj * ((size_t)B_ * T_ * H_);

    const int  strip = blockIdx.x * 8 + wave;       // 16-row strip of x
    const long row   = (long)strip * 16 + l15;      // this lane's A row
    const float* xrow = x + row * D_;

    v8f acc[4] = {};                                // 4 col tiles of 16

    for (int kt = 0; kt < D_ / 32; ++kt) {
        __syncthreads();                            // protect LDS overwrite
        // Stage W[kt*32 .. kt*32+31][0..63] into B-fragment layout (f16 pairs).
        for (int i = 0; i < 4; ++i) {
            int idx = tid + i * 256;                // 0..1023
            int v   = idx & 7;
            int ln  = (idx >> 3) & 31;
            int nt  = idx >> 8;
            int n   = nt * 16 + (ln & 15);
            int kb  = kt * 32 + (v & 3) * 2 + (ln >> 4) * 8 + (v >> 2) * 16;
            union { unsigned u; _Float16 h[2]; } pk;
            pk.h[0] = (_Float16)W[(long)kb * H_ + n];
            pk.h[1] = (_Float16)W[(long)(kb + 1) * H_ + n];
            lds_w[idx] = pk.u;
        }
        __syncthreads();

        // A fragment (16x32 f16): K pairs contiguous in row-major x.
        v16h a;
        for (int v = 0; v < 8; ++v) {
            int kp = (v < 4) ? (kt * 32 + half * 8 + v * 2)
                             : (kt * 32 + 16 + half * 8 + (v - 4) * 2);
            float2 f = *(const float2*)(xrow + kp);
            a[2 * v]     = (_Float16)f.x;
            a[2 * v + 1] = (_Float16)f.y;
        }

        for (int nt = 0; nt < 4; ++nt) {
            const uint4* bp = (const uint4*)&lds_w[(nt * 32 + lane) * 8];
            union { uint4 u4; _Float16 h[8]; } u0, u1;
            u0.u4 = bp[0];
            u1.u4 = bp[1];
            v16h bfrag;
            for (int e = 0; e < 8; ++e) {
                bfrag[e]     = u0.h[e];
                bfrag[8 + e] = u1.h[e];
            }
            acc[nt] = __builtin_amdgcn_wmma_f32_16x16x32_f16(
                false, a, false, bfrag, (short)0, acc[nt], false, false);
        }
    }

    // C layout: lane holds rows M = v + half*8, col = l15 (+nt*16).
    if (proj != 2) {
        for (int nt = 0; nt < 4; ++nt)
            for (int v = 0; v < 8; ++v) {
                long r = (long)strip * 16 + v + half * 8;
                out[r * H_ + nt * 16 + l15] = (_Float16)acc[nt][v];
            }
    } else {
        // V transposed: vT[b][h][t]
        for (int nt = 0; nt < 4; ++nt)
            for (int v = 0; v < 8; ++v) {
                long r  = (long)strip * 16 + v + half * 8;  // global token index
                long bb = r >> 11;                          // / T_
                long t  = r & (T_ - 1);
                long h  = nt * 16 + l15;
                out[(bb * H_ + h) * T_ + t] = (_Float16)acc[nt][v];
            }
    }
}

// ---------------------------------------------------------------------------
// Kernel 2: flash attention, 32 keys per iteration.  One wave = 16 query rows.
// ---------------------------------------------------------------------------
__global__ __launch_bounds__(256) void flash_attn_kernel(
    const _Float16* __restrict__ qf,   // [B][T][H]
    const _Float16* __restrict__ kf,   // [B][T][H]
    const _Float16* __restrict__ vT,   // [B][H][T]
    float* __restrict__ out)           // [B][T][H] fp32
{
    __shared__ _Float16 lds_p[8][16 * 32]; // per-wave 16x32 P tile (C->A relayout)

    const int tid  = threadIdx.x;
    const int wave = tid >> 5;
    const int lane = tid & 31;
    const int half = lane >> 4;
    const int l15  = lane & 15;

    const int g  = blockIdx.x * 8 + wave;  // global query-tile id
    const int b  = g >> 7;                 // T/16 = 128 tiles per batch
    const int qt = g & 127;

    const long base  = (long)b * T_ * H_;
    const long baseT = (long)b * H_ * T_;

    // Q fragments (A layout), two 16x32 K-steps over head dim 64.
    v16h aq[2];
    {
        const unsigned* qrow =
            (const unsigned*)(qf + base + (long)(qt * 16 + l15) * H_);
        for (int s = 0; s < 2; ++s) {
            union { uint4 u4; _Float16 h[8]; } u0, u1;
            u0.u4 = *(const uint4*)(qrow + s * 16 + half * 4);
            u1.u4 = *(const uint4*)(qrow + s * 16 + 8 + half * 4);
            for (int e = 0; e < 8; ++e) {
                aq[s][e]     = u0.h[e];
                aq[s][8 + e] = u1.h[e];
            }
        }
    }

    v8f acc[4] = {};
    float m[8], l[8];
    for (int v = 0; v < 8; ++v) { m[v] = -__builtin_huge_valf(); l[v] = 0.f; }

    const int nkb = (qt >> 1) + 1;         // 32-key blocks
    for (int kb = 0; kb < nkb; ++kb) {
        if (kb + 1 < nkb)                  // prefetch next K block rows
            __builtin_prefetch(
                (const void*)(kf + base + (long)((kb + 1) * 32 + l15) * H_), 0, 1);

        // ---- S = Q*K^T for two 16-key subtiles c=0,1 ----
        v8f s_sub[2];
        for (int c = 0; c < 2; ++c) {
            v8f s_acc = {};
            const unsigned* krow =
                (const unsigned*)(kf + base + (long)(kb * 32 + c * 16 + l15) * H_);
            for (int s = 0; s < 2; ++s) {
                union { uint4 u4; _Float16 h[8]; } u0, u1;
                u0.u4 = *(const uint4*)(krow + s * 16 + half * 4);
                u1.u4 = *(const uint4*)(krow + s * 16 + 8 + half * 4);
                v16h bk;
                for (int e = 0; e < 8; ++e) {
                    bk[e]     = u0.h[e];
                    bk[8 + e] = u1.h[e];
                }
                s_acc = __builtin_amdgcn_wmma_f32_16x16x32_f16(
                    false, aq[s], false, bk, (short)0, s_acc, false, false);
            }
            s_sub[c] = s_acc;
        }

        // ---- scale + causal mask per subtile ----
        float sv[2][8];
        for (int c = 0; c < 2; ++c) {
            const int ktile = kb * 2 + c;
            for (int v = 0; v < 8; ++v) {
                float val = s_sub[c][v] * 0.125f;       // 1/sqrt(64)
                if (ktile > qt || (ktile == qt && l15 > v + half * 8))
                    val = -__builtin_huge_valf();
                sv[c][v] = val;
            }
        }

        // ---- online softmax over the 32 columns ----
        float rm[8];
        for (int v = 0; v < 8; ++v) rm[v] = fmaxf(sv[0][v], sv[1][v]);
        for (int off = 1; off < 16; off <<= 1)
            for (int v = 0; v < 8; ++v)
                rm[v] = fmaxf(rm[v], __shfl_xor(rm[v], off, 32));

        float alpha[8];
        for (int v = 0; v < 8; ++v) {
            float mn = fmaxf(m[v], rm[v]);
            alpha[v] = __expf(m[v] - mn);
            m[v] = mn;
        }

        float p0[8], p1[8], rs[8];
        for (int v = 0; v < 8; ++v) {
            p0[v] = __expf(sv[0][v] - m[v]);
            p1[v] = __expf(sv[1][v] - m[v]);
            rs[v] = p0[v] + p1[v];
        }
        for (int off = 1; off < 16; off <<= 1)
            for (int v = 0; v < 8; ++v)
                rs[v] += __shfl_xor(rs[v], off, 32);
        for (int v = 0; v < 8; ++v) l[v] = l[v] * alpha[v] + rs[v];
        for (int nt = 0; nt < 4; ++nt)
            for (int v = 0; v < 8; ++v)
                acc[nt][v] *= alpha[v];

        // ---- C-layout P (16x32) -> LDS -> A-layout fragment ----
        for (int v = 0; v < 8; ++v) {
            lds_p[wave][(v + half * 8) * 32 + l15]      = (_Float16)p0[v];
            lds_p[wave][(v + half * 8) * 32 + 16 + l15] = (_Float16)p1[v];
        }
        v16h ap;
        {
            union { uint4 u4; _Float16 h[8]; } u0, u1;
            u0.u4 = *(const uint4*)&lds_p[wave][l15 * 32 + half * 8];
            u1.u4 = *(const uint4*)&lds_p[wave][l15 * 32 + 16 + half * 8];
            for (int e = 0; e < 8; ++e) {
                ap[e]     = u0.h[e];
                ap[8 + e] = u1.h[e];
            }
        }

        // ---- O += P * V : vT makes each B-fragment 2x b128 per lane ----
        for (int nt = 0; nt < 4; ++nt) {
            const unsigned* vrow =
                (const unsigned*)(vT + baseT + (long)(nt * 16 + l15) * T_);
            union { uint4 u4; _Float16 h[8]; } u0, u1;
            u0.u4 = *(const uint4*)(vrow + kb * 16 + half * 4);
            u1.u4 = *(const uint4*)(vrow + kb * 16 + 8 + half * 4);
            v16h bv;
            for (int e = 0; e < 8; ++e) {
                bv[e]     = u0.h[e];
                bv[8 + e] = u1.h[e];
            }
            acc[nt] = __builtin_amdgcn_wmma_f32_16x16x32_f16(
                false, ap, false, bv, (short)0, acc[nt], false, false);
        }
    }

    // Epilogue: normalize by row sum, store fp32.
    for (int nt = 0; nt < 4; ++nt)
        for (int v = 0; v < 8; ++v) {
            long r = (long)(qt * 16 + v + half * 8);
            out[base + r * H_ + nt * 16 + l15] = acc[nt][v] / l[v];
        }
}

// ---------------------------------------------------------------------------
extern "C" void kernel_launch(void* const* d_in, const int* in_sizes, int n_in,
                              void* d_out, int out_size, void* d_ws, size_t ws_size,
                              hipStream_t stream) {
    const float* x  = (const float*)d_in[0];
    const float* Wq = (const float*)d_in[1];
    const float* Wk = (const float*)d_in[2];
    const float* Wv = (const float*)d_in[3];
    float* out = (float*)d_out;

    _Float16* qkv = (_Float16*)d_ws;                 // q, k, vT: 3 * 2 MB
    const size_t N = (size_t)B_ * T_ * H_;

    dim3 g1(B_ * T_ / 16 / 8, 3);                    // (128, 3) x 256 threads
    qkv_proj_kernel<<<g1, 256, 0, stream>>>(x, Wq, Wk, Wv, qkv);

    dim3 g2(B_ * T_ / 16 / 8);                       // 128 x 256 threads
    flash_attn_kernel<<<g2, 256, 0, stream>>>(qkv, qkv + N, qkv + 2 * N, out);
}
